// Net_63531156242918
// MI455X (gfx1250) — compile-verified, matrix operand order checked by reading, not run on hardware
//
#include <hip/hip_runtime.h>

// ---------------------------------------------------------------------------
// SNN (4-layer LIF, snntorch-style) fused forward for gfx1250 / MI455X.
//   x: [B=2048, T=32, 48] fp32
//   L1: 48->256, L2/L3: 256->256, L4: 256->10, LIF between each.
// Strategy:
//   * 128 persistent workgroups, each owns 16 batch rows for all 32 steps.
//   * 16 wave32s/WG (512 thr); wave w owns output neurons [16w, 16w+16)
//     (exactly one 16x16 WMMA N-tile) -> per-wave VGPR use stays < 256,
//     avoiding the VGPR-MSB region and its s_set_vgpr_msb overhead,
//     and giving 4 resident waves/SIMD for latency hiding.
//   * Matmuls via v_wmma_f32_16x16x32_f16 (f32 accum), K-chunked by 32.
//   * Weights pre-swizzled once into per-lane B-fragment layout (L2-resident).
//   * Spikes staged through an 8KB LDS buffer, reloaded as A fragments.
//   * Membrane states kept in accumulator-layout VGPRs across timesteps.
// ---------------------------------------------------------------------------

typedef __attribute__((ext_vector_type(16))) _Float16 v16h;
typedef __attribute__((ext_vector_type(8)))  _Float16 v8h;
typedef __attribute__((ext_vector_type(8)))  float    v8f;

#define BATCH   2048
#define TSTEPS  32
#define DIN     48
#define DINP    64      // K of layer 1 padded to 2 chunks of 32
#define HID     256
#define DOUT    10
#define ROWS    16      // batch rows per workgroup (one WMMA M-tile)
#define NTHREADS 512    // 16 wave32s, one N-tile of 16 neurons each

static __device__ __forceinline__ v8f wmma16(v16h a, v16h b, v8f c) {
  // D = A(16x32 f16) * B(32x16 f16) + C(16x16 f32)
  return __builtin_amdgcn_wmma_f32_16x16x32_f16(
      /*neg_a=*/false, a, /*neg_b=*/false, b,
      /*c_mod=*/(short)0, c, /*reuse_a=*/false, /*reuse_b=*/false);
}

// A fragment (16x32, f16) from a row-major [16 x ldk] f16 buffer in LDS.
// ISA layout: lanes 0-15 (M=lane): halves 0..7 = K 0..7, halves 8..15 = K 16..23
//             lanes 16-31 (M=lane-16): halves 0..7 = K 8..15, 8..15 = K 24..31
static __device__ __forceinline__ v16h lds_afrag(const _Float16* S, int ldk,
                                                 int row, int k0, int g) {
  const _Float16* p = S + row * ldk + k0 + g * 8;
  union { v16h v; v8h h[2]; } u;
  u.h[0] = *(const v8h*)(p);        // K: k0 + 8g .. +7
  u.h[1] = *(const v8h*)(p + 16);   // K: k0 + 16 + 8g .. +7
  return u.v;
}

// B fragment: pre-swizzled in global memory, 16 contiguous halves per lane.
static __device__ __forceinline__ v16h gbl_bfrag(const _Float16* p) {
  union { v16h v; v8h h[2]; } u;
  u.h[0] = *(const v8h*)(p);
  u.h[1] = *(const v8h*)(p + 8);
  return u.v;
}

// ---------------------------------------------------------------------------
// Prep: rearrange W [N x K] fp32 into B-fragment order, f16, zero-padded.
// Fragment (n_tile, kc) = 512 halves: lane-major, 16 halves per lane.
// lane L -> column n = n_tile*16 + (L&15), g = L>>4
// half h -> k = kc*32 + (h<8 ? g*8+h : 16 + g*8 + (h-8))
// ---------------------------------------------------------------------------
__global__ void prep_wfrag(const float* __restrict__ W, _Float16* __restrict__ dst,
                           int N, int K, int Kchunks, int Ntiles) {
  int total = Ntiles * Kchunks * 512;
  for (int idx = blockIdx.x * blockDim.x + threadIdx.x; idx < total;
       idx += gridDim.x * blockDim.x) {
    int frag   = idx >> 9;
    int within = idx & 511;
    int lane   = within >> 4;
    int h      = within & 15;
    int n_tile = frag / Kchunks;
    int kc     = frag % Kchunks;
    int n      = n_tile * 16 + (lane & 15);
    int g      = lane >> 4;
    int klocal = (h < 8) ? (g * 8 + h) : (16 + g * 8 + (h - 8));
    int k      = kc * 32 + klocal;
    float v = (n < N && k < K) ? W[(size_t)n * K + k] : 0.0f;
    dst[idx] = (_Float16)v;
  }
}

// ---------------------------------------------------------------------------
// Main fused kernel
// ---------------------------------------------------------------------------
struct LayerOut { float* s; float* m; float* c; };

// Hidden layer (256 outputs): WMMA over K, then LIF update + stores.
// mst = membrane state in accumulator layout for this wave's N-tile (= wid).
static __device__ __forceinline__ void hidden_layer(
    const _Float16* Sin, int ldk, int kchunks,
    const _Float16* __restrict__ wf, const float* __restrict__ bias,
    float beta, float thr, v8f* mst,
    LayerOut o, _Float16* Sout,
    int t, int m0, int wid, int lane) {
  const int g  = lane >> 4;
  const int ln = lane & 15;

  v8f acc = {0.f,0.f,0.f,0.f,0.f,0.f,0.f,0.f};
  for (int kc = 0; kc < kchunks; ++kc) {
    v16h a  = lds_afrag(Sin, ldk, ln, kc * 32, g);
    v16h bb = gbl_bfrag(wf + ((size_t)(wid * kchunks + kc)) * 512 + lane * 16);
    acc = wmma16(a, bb, acc);
  }
  __syncthreads();  // all waves done reading Sin before Sout overwrite

  const int n  = wid * 16 + ln;
  const float bn = bias[n];
  v8f mv = *mst;
#pragma unroll
  for (int r = 0; r < 8; ++r) {
    const int mrow = r + 8 * g;            // local batch row 0..15
    float cv    = acc[r] + bn;
    float mp    = mv[r];
    float reset = (mp > thr) ? 1.0f : 0.0f;       // heaviside(mem_prev - thr)
    float mn    = beta * mp + cv - reset * thr;
    float sp    = (mn > thr) ? 1.0f : 0.0f;       // heaviside(mem_new - thr)
    mv[r] = mn;
    size_t idx = ((size_t)t * BATCH + (m0 + mrow)) * HID + n;
    o.c[idx] = cv;
    o.m[idx] = mn;
    o.s[idx] = sp;
    Sout[mrow * HID + n] = (_Float16)sp;
  }
  *mst = mv;
  __syncthreads();  // Sout fully staged for next layer
}

__global__ __launch_bounds__(NTHREADS) void snn_fused(
    const float* __restrict__ x,
    const _Float16* __restrict__ wf1, const _Float16* __restrict__ wf2,
    const _Float16* __restrict__ wf3, const _Float16* __restrict__ wf4,
    const float* __restrict__ b1, const float* __restrict__ b2,
    const float* __restrict__ b3, const float* __restrict__ b4,
    const float* __restrict__ pbe1, const float* __restrict__ pbe2,
    const float* __restrict__ pbe3, const float* __restrict__ pbe4,
    const float* __restrict__ pth1, const float* __restrict__ pth2,
    const float* __restrict__ pth3, const float* __restrict__ pth4,
    float* __restrict__ out) {
  constexpr size_t TBH = (size_t)TSTEPS * BATCH * HID;
  constexpr size_t TBD = (size_t)TSTEPS * BATCH * DOUT;
  LayerOut o1 = { out + 0 * TBH, out + 3 * TBH + 1 * TBD, out + 6 * TBH + 2 * TBD };
  LayerOut o2 = { out + 1 * TBH, out + 4 * TBH + 1 * TBD, out + 7 * TBH + 2 * TBD };
  LayerOut o3 = { out + 2 * TBH, out + 5 * TBH + 1 * TBD, out + 8 * TBH + 2 * TBD };
  LayerOut o4 = { out + 3 * TBH, out + 6 * TBH + 1 * TBD, out + 9 * TBH + 2 * TBD };

  __shared__ __align__(32) _Float16 Ssp[ROWS * HID];   // spike staging (8 KB)
  __shared__ __align__(32) _Float16 Xsh[ROWS * DINP];  // x_t staging   (2 KB)

  const int tid  = threadIdx.x;
  const int lane = tid & 31;
  const int wid  = tid >> 5;           // 0..15 -> N-tile index
  const int g    = lane >> 4;
  const int ln   = lane & 15;
  const int m0   = blockIdx.x * ROWS;  // batch row base for this WG

  // clip(beta, 0, 1) once (beta is constant over time)
  const float be1 = fminf(fmaxf(*pbe1, 0.f), 1.f);
  const float be2 = fminf(fmaxf(*pbe2, 0.f), 1.f);
  const float be3 = fminf(fmaxf(*pbe3, 0.f), 1.f);
  const float be4 = fminf(fmaxf(*pbe4, 0.f), 1.f);
  const float th1 = *pth1, th2 = *pth2, th3 = *pth3, th4 = *pth4;

  // Membrane state in accumulator layout, persistent over t.
  v8f zero = {0.f,0.f,0.f,0.f,0.f,0.f,0.f,0.f};
  v8f mst1 = zero;
  v8f mst2 = zero;
  v8f mst3 = zero;
  v8f mst4 = zero;                     // wave 0 only (N-tile 0, cols 0..15)

  for (int t = 0; t < TSTEPS; ++t) {
    // Stage x[:, t, :] into LDS as f16, K padded 48 -> 64 with zeros.
    for (int i = tid; i < ROWS * DINP; i += NTHREADS) {
      int r = i >> 6, k = i & (DINP - 1);
      float v = (k < DIN) ? x[((size_t)(m0 + r) * TSTEPS + t) * DIN + k] : 0.0f;
      Xsh[i] = (_Float16)v;
    }
    __syncthreads();  // X staged; also fences wave0's layer-4 reads of Ssp (prev t)

    // Layer 1: 48(->64) -> 256
    hidden_layer(Xsh, DINP, 2, wf1, b1, be1, th1, &mst1, o1, Ssp, t, m0, wid, lane);
    // Layer 2: 256 -> 256
    hidden_layer(Ssp, HID, 8, wf2, b2, be2, th2, &mst2, o2, Ssp, t, m0, wid, lane);
    // Layer 3: 256 -> 256
    hidden_layer(Ssp, HID, 8, wf3, b3, be3, th3, &mst3, o3, Ssp, t, m0, wid, lane);

    // Layer 4: 256 -> 10 (N padded to 16); wave-uniform branch keeps EXEC full.
    if (wid == 0) {
      v8f acc = zero;
      for (int kc = 0; kc < 8; ++kc) {
        v16h a  = lds_afrag(Ssp, HID, ln, kc * 32, g);
        v16h bb = gbl_bfrag(wf4 + (size_t)kc * 512 + lane * 16);
        acc = wmma16(a, bb, acc);
      }
      const float bn = (ln < DOUT) ? b4[ln] : 0.0f;
#pragma unroll
      for (int r = 0; r < 8; ++r) {
        const int mrow = r + 8 * g;
        float cv    = acc[r] + bn;
        float mp    = mst4[r];
        float reset = (mp > th4) ? 1.0f : 0.0f;
        float mn    = be4 * mp + cv - reset * th4;
        float sp    = (mn > th4) ? 1.0f : 0.0f;
        mst4[r] = mn;
        if (ln < DOUT) {
          size_t idx = ((size_t)t * BATCH + (m0 + mrow)) * DOUT + ln;
          o4.c[idx] = cv;
          o4.m[idx] = mn;
          o4.s[idx] = sp;
        }
      }
    }
    // Ssp (s3) reads by wave 0 are fenced by the sync at the top of next t.
  }
}

// ---------------------------------------------------------------------------
// Host launch
// ---------------------------------------------------------------------------
extern "C" void kernel_launch(void* const* d_in, const int* in_sizes, int n_in,
                              void* d_out, int out_size, void* d_ws, size_t ws_size,
                              hipStream_t stream) {
  (void)in_sizes; (void)n_in; (void)out_size; (void)ws_size;

  const float* x  = (const float*)d_in[0];
  const float* W1 = (const float*)d_in[1];
  const float* b1 = (const float*)d_in[2];
  const float* W2 = (const float*)d_in[3];
  const float* b2 = (const float*)d_in[4];
  const float* W3 = (const float*)d_in[5];
  const float* b3 = (const float*)d_in[6];
  const float* W4 = (const float*)d_in[7];
  const float* b4 = (const float*)d_in[8];
  const float* be1 = (const float*)d_in[9];
  const float* be2 = (const float*)d_in[10];
  const float* be3 = (const float*)d_in[11];
  const float* be4 = (const float*)d_in[12];
  const float* th1 = (const float*)d_in[13];
  const float* th2 = (const float*)d_in[14];
  const float* th3 = (const float*)d_in[15];
  const float* th4 = (const float*)d_in[16];

  // Workspace layout (f16 halves): pre-swizzled B fragments.
  _Float16* wf1 = (_Float16*)d_ws;                  // 16 tiles * 2 chunks * 512
  _Float16* wf2 = wf1 + 16 * 2 * 512;               // 16 * 8 * 512
  _Float16* wf3 = wf2 + 16 * 8 * 512;               // 16 * 8 * 512
  _Float16* wf4 = wf3 + 16 * 8 * 512;               //  1 * 8 * 512

  prep_wfrag<<<(16 * 2 * 512 + 255) / 256, 256, 0, stream>>>(W1, wf1, HID, DIN, 2, 16);
  prep_wfrag<<<(16 * 8 * 512 + 255) / 256, 256, 0, stream>>>(W2, wf2, HID, HID, 8, 16);
  prep_wfrag<<<(16 * 8 * 512 + 255) / 256, 256, 0, stream>>>(W3, wf3, HID, HID, 8, 16);
  prep_wfrag<<<( 1 * 8 * 512 + 255) / 256, 256, 0, stream>>>(W4, wf4, DOUT, HID, 8, 1);

  snn_fused<<<BATCH / ROWS, NTHREADS, 0, stream>>>(
      x, wf1, wf2, wf3, wf4, b1, b2, b3, b4,
      be1, be2, be3, be4, th1, th2, th3, th4, (float*)d_out);
}
